// PointNet2Regressor_3Dbbox_15324443312193
// MI455X (gfx1250) — matrix operand order repaired and checked by reading, not compiled, not run on hardware
//
#include <hip/hip_runtime.h>

// ---------------------------------------------------------------------------
// Types for gfx1250 WMMA (wave32). bf16 inputs, f32 accumulate.
// ---------------------------------------------------------------------------
typedef __bf16 bf16_t;
typedef __attribute__((ext_vector_type(8)))  bf16_t v8bf;
typedef __attribute__((ext_vector_type(16))) bf16_t v16bf;
typedef __attribute__((ext_vector_type(8)))  float  v8f;
typedef __attribute__((ext_vector_type(4)))  int    v4i;

#define BN_SCALE_F 0.999995000037499f   // 1/sqrt(1+1e-5)

__device__ __forceinline__ bf16_t f2bf(float f) {
  union { float f; unsigned int u; } cv; cv.f = f;
  unsigned int u = cv.u;
  u += 0x7FFFu + ((u >> 16) & 1u);          // round-to-nearest-even
  unsigned short s = (unsigned short)(u >> 16);
  bf16_t r;
  __builtin_memcpy(&r, &s, sizeof(r));
  return r;
}
__device__ __forceinline__ float bf2f(bf16_t h) {
  unsigned short s;
  __builtin_memcpy(&s, &h, sizeof(s));
  union { unsigned int u; float f; } cv; cv.u = ((unsigned int)s) << 16;
  return cv.f;
}

// ---------------------------------------------------------------------------
// CDNA5 async copy global -> LDS (ASYNCcnt path), with compile-safe fallback.
// Builtin signature (from hipcc diagnostic): (v4i AS1*, v4i AS3*, imm, imm).
// On gfx1250, AS3 pointers are 32-bit and the LDS aperture maps flat addr[31:0]
// to the wave's LDS offset (ISA 10.2). memcpy of sizeof(ptr) bytes from the
// flat address is size-correct on BOTH host (8B) and device (4B) passes.
// ---------------------------------------------------------------------------
#define ASYNC_LDS_OK __has_builtin(__builtin_amdgcn_global_load_async_to_lds_b128)

typedef __attribute__((address_space(1))) v4i* g_v4i_p;
typedef __attribute__((address_space(3))) v4i* l_v4i_p;

__device__ __forceinline__ l_v4i_p to_lds_ptr(void* p) {
  size_t v = (size_t)p;
  l_v4i_p r;
  __builtin_memcpy(&r, &v, sizeof(r));   // low bytes of flat addr = LDS offset
  return r;
}
__device__ __forceinline__ g_v4i_p to_gbl_ptr(const void* p) {
  size_t v = (size_t)p;
  g_v4i_p r;
  __builtin_memcpy(&r, &v, sizeof(r));
  return r;
}

__device__ __forceinline__ void stage16(const bf16_t* g, bf16_t* l) {
#if ASYNC_LDS_OK
  __builtin_amdgcn_global_load_async_to_lds_b128(to_gbl_ptr(g), to_lds_ptr(l), 0, 0);
#else
  *(v8bf*)l = *(const v8bf*)g;     // fallback: global_load_b128 + ds_store_b128
#endif
}
__device__ __forceinline__ void wait_async_le1() {
#if ASYNC_LDS_OK
#if __has_builtin(__builtin_amdgcn_s_wait_asynccnt)
  __builtin_amdgcn_s_wait_asynccnt(1);
#else
  asm volatile("s_wait_asynccnt 1" ::: "memory");
#endif
#endif
}
__device__ __forceinline__ void wait_async_le0() {
#if ASYNC_LDS_OK
#if __has_builtin(__builtin_amdgcn_s_wait_asynccnt)
  __builtin_amdgcn_s_wait_asynccnt(0);
#else
  asm volatile("s_wait_asynccnt 0" ::: "memory");
#endif
#endif
}

// ---------------------------------------------------------------------------
// Generic WMMA GEMM:  out[r][o] = act( (sum_k A[r][k]*W[o][k] + b[o]) * g' + beta )
// A: rows x K (bf16, row-major, K mult of 32, rows mult of 32)
// W: Cout x K (bf16, row-major, zero-padded)
// Block = 256 threads = 8 waves. Block tile: 256 rows x 64 cols.
// Wave tile: 32 rows (2 A fragments) x 64 cols (4 B fragments) = 8 WMMA / K-tile.
// Weight K-tile (64 cols x 32 K = 4KB) staged in LDS via async copy,
// double-buffered so the copy of tile k+1 overlaps WMMAs of tile k.
// flags: bit0 = BN+ReLU, bit1 = write bf16, bit2 = write f32
// ---------------------------------------------------------------------------
__global__ __launch_bounds__(256)
void wmma_gemm_kernel(const bf16_t* __restrict__ A, const bf16_t* __restrict__ W,
                      const float* __restrict__ bias, const float* __restrict__ gamma,
                      const float* __restrict__ beta,
                      bf16_t* __restrict__ outB, float* __restrict__ outF,
                      int rows, int K, int Cout, int CoutValid, int flags)
{
  __shared__ bf16_t wtile[2][64 * 32];   // [buf][col][k]  (2 x 4KB)

  const int tid   = threadIdx.x;
  const int lane  = tid & 31;
  const int wave  = tid >> 5;
  const int nbase = blockIdx.y * 64;
  const int half  = lane >> 4;           // K-half selector
  const int lrow  = lane & 15;           // A row / B col within 16-tile

  const int mbase  = (blockIdx.x * 8 + wave) * 32;
  const bool active = mbase < rows;      // inactive waves still join barriers
  const int m0 = active ? mbase : 0;

  // Staging role: 256 threads x 16B cover the 64x32 bf16 tile.
  const int so = tid >> 2;               // 0..63 : output channel within block tile
  const int sc = tid & 3;                // 0..3  : 16B chunk within 64B row
  const bf16_t* wsrc = W + (size_t)(nbase + so) * K + sc * 8;
  bf16_t* wdst[2] = { &wtile[0][so * 32 + sc * 8], &wtile[1][so * 32 + sc * 8] };

  // A fragments (16-bit A 16x32 layout): lane holds K = half*8+[0..7] and 16+half*8+[0..7]
  const bf16_t* aptr0 = A + (size_t)(m0 + lrow) * K + (half << 3);
  const bf16_t* aptr1 = aptr0 + (size_t)16 * K;

  v8f acc[8] = {};                       // [mt*4 + t]
  const int ktiles = K >> 5;

  stage16(wsrc, wdst[0]);                // preload K-tile 0

  for (int kt = 0; kt < ktiles; ++kt) {
    const int buf = kt & 1;
    if (kt + 1 < ktiles) {
      stage16(wsrc + (size_t)(kt + 1) * 32, wdst[(kt + 1) & 1]);
      wait_async_le1();                  // K-tile kt complete (in-order)
    } else {
      wait_async_le0();
    }
    __syncthreads();                     // staged tile visible to all waves

    const bf16_t* ap0 = aptr0 + (kt << 5);
    const bf16_t* ap1 = aptr1 + (kt << 5);
    v8bf a0lo = *(const v8bf*)(ap0);
    v8bf a0hi = *(const v8bf*)(ap0 + 16);
    v8bf a1lo = *(const v8bf*)(ap1);
    v8bf a1hi = *(const v8bf*)(ap1 + 16);
    v16bf a0 = __builtin_shufflevector(a0lo, a0hi,
        0,1,2,3,4,5,6,7,8,9,10,11,12,13,14,15);
    v16bf a1 = __builtin_shufflevector(a1lo, a1hi,
        0,1,2,3,4,5,6,7,8,9,10,11,12,13,14,15);
    __builtin_prefetch((const void*)(ap0 + 32), 0, 1);
    __builtin_prefetch((const void*)(ap1 + 32), 0, 1);

#pragma unroll
    for (int t = 0; t < 4; ++t) {
      // B fragment from LDS: lane = col, K = half*8+[0..7] and 16+half*8+[0..7]
      const bf16_t* bp = &wtile[buf][(t * 16 + lrow) * 32 + (half << 3)];
      v16bf bfrag = __builtin_shufflevector(*(const v8bf*)bp, *(const v8bf*)(bp + 16),
          0,1,2,3,4,5,6,7,8,9,10,11,12,13,14,15);   // ds_load_b128 x2
      acc[t]     = __builtin_amdgcn_wmma_f32_16x16x32_bf16(
          false, a0, false, bfrag, (short)0, acc[t],     false, false);
      acc[4 + t] = __builtin_amdgcn_wmma_f32_16x16x32_bf16(
          false, a1, false, bfrag, (short)0, acc[4 + t], false, false);
    }
    __syncthreads();                     // all reads done before buf is re-staged
  }

  const bool do_bn = (flags & 1) != 0;
  const bool wrB   = (flags & 2) != 0;
  const bool wrF   = (flags & 4) != 0;

#pragma unroll
  for (int t = 0; t < 4; ++t) {
    const int o = nbase + (t << 4) + lrow;
    if (active && o < CoutValid) {
      const float bb = bias[o];
      const float sc_ = do_bn ? gamma[o] * BN_SCALE_F : 1.0f;
      const float bt  = do_bn ? beta[o] : 0.0f;
#pragma unroll
      for (int mt = 0; mt < 2; ++mt) {
#pragma unroll
        for (int j = 0; j < 8; ++j) {
          // C/D layout: VGPR j holds row M = half*8 + j, col = lane&15
          const int r = m0 + mt * 16 + (half << 3) + j;
          float v = acc[mt * 4 + t][j] + bb;
          v = v * sc_ + bt;
          if (do_bn) v = fmaxf(v, 0.0f);
          const size_t off = (size_t)r * Cout + o;
          if (wrB) outB[off] = f2bf(v);
          if (wrF) outF[off] = v;
        }
      }
    }
  }
}

// ---------------------------------------------------------------------------
// (B,3,N) -> (B,N,3)
// ---------------------------------------------------------------------------
__global__ void transpose_xyz_kernel(const float* __restrict__ in,
                                     float* __restrict__ out, int Bc, int Nc)
{
  int i = blockIdx.x * 256 + threadIdx.x;
  if (i >= Bc * Nc) return;
  int b = i / Nc, p = i % Nc;
  out[(size_t)i * 3 + 0] = in[((size_t)b * 3 + 0) * Nc + p];
  out[(size_t)i * 3 + 1] = in[((size_t)b * 3 + 1) * Nc + p];
  out[(size_t)i * 3 + 2] = in[((size_t)b * 3 + 2) * Nc + p];
}

// ---------------------------------------------------------------------------
// fp32 weights (Cout x Cin) -> bf16 (CoutPad x Kpad), zero padded
// ---------------------------------------------------------------------------
__global__ void convert_w_kernel(const float* __restrict__ w, bf16_t* __restrict__ out,
                                 int Cout, int Cin, int CoutPad, int Kpad)
{
  int i = blockIdx.x * 256 + threadIdx.x;
  if (i >= CoutPad * Kpad) return;
  int o = i / Kpad, c = i % Kpad;
  float v = (o < Cout && c < Cin) ? w[(size_t)o * Cin + c] : 0.0f;
  out[i] = f2bf(v);
}

// ---------------------------------------------------------------------------
// Farthest point sampling: one block per batch, serial npoint iterations.
// Points + running distances cached in LDS; argmax via LDS tree reduction.
// ---------------------------------------------------------------------------
__global__ __launch_bounds__(256)
void fps_kernel(const float* __restrict__ pts, int n, int npoint,
                int* __restrict__ fps_idx, float* __restrict__ new_xyz)
{
  __shared__ float sx[2048], sy[2048], sz[2048], sd[2048];
  __shared__ float rv[256];
  __shared__ int   ri[256];

  const int b = blockIdx.x;
  const float* base = pts + (size_t)b * n * 3;
  for (int i = threadIdx.x; i < n; i += 256) {
    sx[i] = base[(size_t)i * 3 + 0];
    sy[i] = base[(size_t)i * 3 + 1];
    sz[i] = base[(size_t)i * 3 + 2];
    sd[i] = 1.0e10f;
  }
  __syncthreads();

  int cur = 0;
  for (int t = 0; t < npoint; ++t) {
    if (threadIdx.x == 0) {
      fps_idx[(size_t)b * npoint + t] = cur;
      new_xyz[((size_t)b * npoint + t) * 3 + 0] = sx[cur];
      new_xyz[((size_t)b * npoint + t) * 3 + 1] = sy[cur];
      new_xyz[((size_t)b * npoint + t) * 3 + 2] = sz[cur];
    }
    const float cx = sx[cur], cy = sy[cur], cz = sz[cur];
    float bestv = -1.0f; int besti = 0;
    for (int i = threadIdx.x; i < n; i += 256) {
      float dx = sx[i] - cx, dy = sy[i] - cy, dz = sz[i] - cz;
      float d  = dx * dx + dy * dy + dz * dz;
      float nd = fminf(sd[i], d);
      sd[i] = nd;
      if (nd > bestv) { bestv = nd; besti = i; }
    }
    rv[threadIdx.x] = bestv; ri[threadIdx.x] = besti;
    __syncthreads();
    for (int s = 128; s > 0; s >>= 1) {
      if ((int)threadIdx.x < s) {
        float ov = rv[threadIdx.x + s]; int oi = ri[threadIdx.x + s];
        if (ov > rv[threadIdx.x] || (ov == rv[threadIdx.x] && oi < ri[threadIdx.x])) {
          rv[threadIdx.x] = ov; ri[threadIdx.x] = oi;
        }
      }
      __syncthreads();
    }
    cur = ri[0];
    __syncthreads();
  }
}

// ---------------------------------------------------------------------------
// Ball query: first `ns` indices (ascending) within radius; pad with first.
// ---------------------------------------------------------------------------
__global__ void ballquery_kernel(const float* __restrict__ pts,
                                 const float* __restrict__ query,
                                 int n, int s, float r2, int ns,
                                 int* __restrict__ idx, int count)
{
  int q = blockIdx.x * 256 + threadIdx.x;
  if (q >= count) return;
  int b = q / s;
  const float* base = pts + (size_t)b * n * 3;
  const float qx = query[(size_t)q * 3 + 0];
  const float qy = query[(size_t)q * 3 + 1];
  const float qz = query[(size_t)q * 3 + 2];
  int* out = idx + (size_t)q * ns;
  int cnt = 0;
  for (int j = 0; j < n && cnt < ns; ++j) {
    float dx = base[(size_t)j * 3 + 0] - qx;
    float dy = base[(size_t)j * 3 + 1] - qy;
    float dz = base[(size_t)j * 3 + 2] - qz;
    if (dx * dx + dy * dy + dz * dz <= r2) out[cnt++] = j;
  }
  int first = (cnt > 0) ? out[0] : 0;
  for (; cnt < ns; ++cnt) out[cnt] = first;
}

// ---------------------------------------------------------------------------
// Grouping: row = (b*s + q)*ns + k -> [xyz diff (3) | feat (C) | zero pad]
// ---------------------------------------------------------------------------
__global__ void group_feat_kernel(const float* __restrict__ pts,
                                  const float* __restrict__ new_xyz,
                                  const int* __restrict__ idx,
                                  const float* __restrict__ feat, int C,
                                  bf16_t* __restrict__ out,
                                  int n, int s, int ns, int Kpad, int count)
{
  int r = blockIdx.x * 256 + threadIdx.x;
  if (r >= count) return;
  int k = r % ns;
  int q = r / ns;
  int b = q / s;
  int j = idx[(size_t)q * ns + k];
  const float* p = pts + ((size_t)b * n + j) * 3;
  const float* c = new_xyz + (size_t)q * 3;
  bf16_t* o = out + (size_t)r * Kpad;
  o[0] = f2bf(p[0] - c[0]);
  o[1] = f2bf(p[1] - c[1]);
  o[2] = f2bf(p[2] - c[2]);
  int t = 3;
  if (C > 0) {
    const float* f = feat + ((size_t)b * n + j) * C;
    for (int cc = 0; cc < C; ++cc) o[t++] = f2bf(f[cc]);
  }
  for (; t < Kpad; ++t) o[t] = f2bf(0.0f);
}

// group_all (SA3): row r in [0, B*n): [xyz (3) | feat (C) | pad]
__global__ void group_all_kernel(const float* __restrict__ xyz,
                                 const float* __restrict__ feat, int C,
                                 bf16_t* __restrict__ out, int Kpad, int count)
{
  int r = blockIdx.x * 256 + threadIdx.x;
  if (r >= count) return;
  const float* p = xyz + (size_t)r * 3;
  const float* f = feat + (size_t)r * C;
  bf16_t* o = out + (size_t)r * Kpad;
  o[0] = f2bf(p[0]); o[1] = f2bf(p[1]); o[2] = f2bf(p[2]);
  int t = 3;
  for (int cc = 0; cc < C; ++cc) o[t++] = f2bf(f[cc]);
  for (; t < Kpad; ++t) o[t] = f2bf(0.0f);
}

// ---------------------------------------------------------------------------
// Max-pool over the neighbor axis: (groups, ns, C) bf16 -> (groups, C) f32
// ---------------------------------------------------------------------------
__global__ void maxpool_kernel(const bf16_t* __restrict__ in, float* __restrict__ out,
                               int groups, int ns, int C)
{
  int i = blockIdx.x * 256 + threadIdx.x;
  if (i >= groups * C) return;
  int g = i / C, c = i % C;
  const bf16_t* p = in + ((size_t)g * ns) * C + c;
  float m = -1.0e30f;
  for (int k = 0; k < ns; ++k) m = fmaxf(m, bf2f(p[(size_t)k * C]));
  out[i] = m;
}

__global__ void f32_to_bf16_kernel(const float* __restrict__ in,
                                   bf16_t* __restrict__ out, int count)
{
  int i = blockIdx.x * 256 + threadIdx.x;
  if (i >= count) return;
  out[i] = f2bf(in[i]);
}

__global__ void final_copy_kernel(const float* __restrict__ in, float* __restrict__ out,
                                  int bc, int validC, int padC)
{
  int i = blockIdx.x * 256 + threadIdx.x;
  if (i >= bc * validC) return;
  int b = i / validC, j = i % validC;
  out[i] = in[(size_t)b * padC + j];
}

// ---------------------------------------------------------------------------
// Host orchestration (graph-capture safe: launches only).
// Input order (setup_inputs dict order, depth-first):
//   0: xyz
//   1..36: sa1/sa2/sa3, per layer {w,b,g,beta} x 9 layers
//   37,38: fc1 w,b   39,40: fc2 w,b   41,42: fc3 w,b
//   43,44: bn1 g,beta   45,46: bn2 g,beta
// ---------------------------------------------------------------------------
extern "C" void kernel_launch(void* const* d_in, const int* in_sizes, int n_in,
                              void* d_out, int out_size, void* d_ws, size_t ws_size,
                              hipStream_t stream)
{
  (void)in_sizes; (void)n_in; (void)out_size; (void)ws_size;

  const float* xyz = (const float*)d_in[0];
  const float *sw[9], *sb[9], *sg[9], *sbe[9];
  int pi = 1;
  for (int l = 0; l < 9; ++l) {
    sw[l]  = (const float*)d_in[pi++];
    sb[l]  = (const float*)d_in[pi++];
    sg[l]  = (const float*)d_in[pi++];
    sbe[l] = (const float*)d_in[pi++];
  }
  const float* fc1w = (const float*)d_in[37]; const float* fc1b = (const float*)d_in[38];
  const float* fc2w = (const float*)d_in[39]; const float* fc2b = (const float*)d_in[40];
  const float* fc3w = (const float*)d_in[41]; const float* fc3b = (const float*)d_in[42];
  const float* bn1g = (const float*)d_in[43]; const float* bn1b = (const float*)d_in[44];
  const float* bn2g = (const float*)d_in[45]; const float* bn2b = (const float*)d_in[46];

  const int B = 32, N = 2048, S1 = 512, NS1 = 32, S2 = 128, NS2 = 64;

  char* ws = (char*)d_ws;
  size_t off = 0;
  auto alloc = [&](size_t bytes) -> void* {
    void* p = ws + off;
    off += (bytes + 255) & ~(size_t)255;
    return p;
  };

  float* pts = (float*)alloc((size_t)B * N * 3 * 4);

  // SA MLP dims: Cin (true), Cout, Kpad (Cin padded to mult of 32)
  const int Cin_[9]  = {3, 64, 64,   131, 128, 128,   259, 256, 512};
  const int Cout_[9] = {64, 64, 128, 128, 128, 256,   256, 512, 1024};
  const int Kpad_[9] = {32, 64, 64,  160, 128, 128,   288, 256, 512};
  bf16_t* wsa[9];
  for (int l = 0; l < 9; ++l)
    wsa[l] = (bf16_t*)alloc((size_t)Cout_[l] * Kpad_[l] * 2);
  bf16_t* wfc1 = (bf16_t*)alloc((size_t)1024 * 1024 * 2);
  bf16_t* wfc2 = (bf16_t*)alloc((size_t)1024 * 1024 * 2);
  bf16_t* wfc3 = (bf16_t*)alloc((size_t)64 * 1024 * 2);

  int*   fpsidx1 = (int*)alloc((size_t)B * S1 * 4);
  float* nxyz1   = (float*)alloc((size_t)B * S1 * 3 * 4);
  int*   idx1    = (int*)alloc((size_t)B * S1 * NS1 * 4);
  int*   fpsidx2 = (int*)alloc((size_t)B * S2 * 4);
  float* nxyz2   = (float*)alloc((size_t)B * S2 * 3 * 4);
  int*   idx2    = (int*)alloc((size_t)B * S2 * NS2 * 4);

  float* pooled1 = (float*)alloc((size_t)B * S1 * 128 * 4);
  float* pooled2 = (float*)alloc((size_t)B * S2 * 256 * 4);
  float* pooled3 = (float*)alloc((size_t)B * 1024 * 4);

  bf16_t* fcin   = (bf16_t*)alloc((size_t)B * 1024 * 2);
  bf16_t* fcact1 = (bf16_t*)alloc((size_t)B * 1024 * 2);
  bf16_t* fcact2 = (bf16_t*)alloc((size_t)B * 1024 * 2);
  float*  fcoutF = (float*)alloc((size_t)B * 64 * 4);

  bf16_t* bufA = (bf16_t*)alloc((size_t)262144 * 160 * 2);  // 80 MB
  bf16_t* bufB = (bf16_t*)alloc((size_t)524288 * 128 * 2);  // 128 MB

  auto blocks = [](int count) { return (count + 255) / 256; };

  auto gemm = [&](const bf16_t* Ain, const bf16_t* Wm, const float* b_,
                  const float* g_, const float* be_, bf16_t* oB, float* oF,
                  int rows, int K, int Cout, int CoutV, int flags) {
    dim3 grid((rows + 255) / 256, Cout / 64);
    wmma_gemm_kernel<<<grid, 256, 0, stream>>>(Ain, Wm, b_, g_, be_, oB, oF,
                                               rows, K, Cout, CoutV, flags);
  };

  // ---- prep ----
  transpose_xyz_kernel<<<blocks(B * N), 256, 0, stream>>>(xyz, pts, B, N);
  for (int l = 0; l < 9; ++l)
    convert_w_kernel<<<blocks(Cout_[l] * Kpad_[l]), 256, 0, stream>>>(
        sw[l], wsa[l], Cout_[l], Cin_[l], Cout_[l], Kpad_[l]);
  convert_w_kernel<<<blocks(1024 * 1024), 256, 0, stream>>>(fc1w, wfc1, 1024, 1024, 1024, 1024);
  convert_w_kernel<<<blocks(1024 * 1024), 256, 0, stream>>>(fc2w, wfc2, 1024, 1024, 1024, 1024);
  convert_w_kernel<<<blocks(64 * 1024), 256, 0, stream>>>(fc3w, wfc3, 60, 1024, 64, 1024);

  // ---- SA1 ----
  fps_kernel<<<B, 256, 0, stream>>>(pts, N, S1, fpsidx1, nxyz1);
  ballquery_kernel<<<blocks(B * S1), 256, 0, stream>>>(pts, nxyz1, N, S1, 0.04f, NS1,
                                                       idx1, B * S1);
  group_feat_kernel<<<blocks(B * S1 * NS1), 256, 0, stream>>>(
      pts, nxyz1, idx1, (const float*)nullptr, 0, bufA, N, S1, NS1, 32, B * S1 * NS1);
  gemm(bufA, wsa[0], sb[0], sg[0], sbe[0], bufB, nullptr, B * S1 * NS1, 32, 64, 64, 1 | 2);
  gemm(bufB, wsa[1], sb[1], sg[1], sbe[1], bufA, nullptr, B * S1 * NS1, 64, 64, 64, 1 | 2);
  gemm(bufA, wsa[2], sb[2], sg[2], sbe[2], bufB, nullptr, B * S1 * NS1, 64, 128, 128, 1 | 2);
  maxpool_kernel<<<blocks(B * S1 * 128), 256, 0, stream>>>(bufB, pooled1, B * S1, NS1, 128);

  // ---- SA2 ----
  fps_kernel<<<B, 256, 0, stream>>>(nxyz1, S1, S2, fpsidx2, nxyz2);
  ballquery_kernel<<<blocks(B * S2), 256, 0, stream>>>(nxyz1, nxyz2, S1, S2, 0.16f, NS2,
                                                       idx2, B * S2);
  group_feat_kernel<<<blocks(B * S2 * NS2), 256, 0, stream>>>(
      nxyz1, nxyz2, idx2, pooled1, 128, bufA, S1, S2, NS2, 160, B * S2 * NS2);
  gemm(bufA, wsa[3], sb[3], sg[3], sbe[3], bufB, nullptr, B * S2 * NS2, 160, 128, 128, 1 | 2);
  gemm(bufB, wsa[4], sb[4], sg[4], sbe[4], bufA, nullptr, B * S2 * NS2, 128, 128, 128, 1 | 2);
  gemm(bufA, wsa[5], sb[5], sg[5], sbe[5], bufB, nullptr, B * S2 * NS2, 128, 256, 256, 1 | 2);
  maxpool_kernel<<<blocks(B * S2 * 256), 256, 0, stream>>>(bufB, pooled2, B * S2, NS2, 256);

  // ---- SA3 (group_all) ----
  group_all_kernel<<<blocks(B * S2), 256, 0, stream>>>(nxyz2, pooled2, 256, bufA, 288, B * S2);
  gemm(bufA, wsa[6], sb[6], sg[6], sbe[6], bufB, nullptr, B * S2, 288, 256, 256, 1 | 2);
  gemm(bufB, wsa[7], sb[7], sg[7], sbe[7], bufA, nullptr, B * S2, 256, 512, 512, 1 | 2);
  gemm(bufA, wsa[8], sb[8], sg[8], sbe[8], bufB, nullptr, B * S2, 512, 1024, 1024, 1 | 2);
  maxpool_kernel<<<blocks(B * 1024), 256, 0, stream>>>(bufB, pooled3, B, S2, 1024);

  // ---- FC head ----
  f32_to_bf16_kernel<<<blocks(B * 1024), 256, 0, stream>>>(pooled3, fcin, B * 1024);
  gemm(fcin,   wfc1, fc1b, bn1g, bn1b, fcact1, nullptr, B, 1024, 1024, 1024, 1 | 2);
  gemm(fcact1, wfc2, fc2b, bn2g, bn2b, fcact2, nullptr, B, 1024, 1024, 1024, 1 | 2);
  gemm(fcact2, wfc3, fc3b, nullptr, nullptr, nullptr, fcoutF, B, 1024, 64, 60, 4);
  final_copy_kernel<<<blocks(B * 60), 256, 0, stream>>>(fcoutF, (float*)d_out, B, 60, 64);
}
